// VectorQuantizer_24215025615106
// MI455X (gfx1250) — matrix operand order repaired and checked by reading, not compiled, not run on hardware
//
#include <hip/hip_runtime.h>
#include <hip/hip_bf16.h>

typedef _Float16 half8  __attribute__((ext_vector_type(8)));
typedef _Float16 half16 __attribute__((ext_vector_type(16)));
typedef float    float8 __attribute__((ext_vector_type(8)));

#define N_ROWS 32768   // 32*32*32
#define DIMS   256
#define KCB    1024
#define HW     1024    // 32*32
#define BATCH  32

// ---------------------------------------------------------------------------
// Zero the histogram + SSE accumulator (must be re-zeroed every call).
__global__ void vq_init(float* __restrict__ hist, float* __restrict__ sse) {
    int t = threadIdx.x;
    if (t < KCB) hist[t] = 0.0f;
    if (t == 0) *sse = 0.0f;
}

// ---------------------------------------------------------------------------
// Codebook: f32 [K][256] -> f16 row-major + cnorm[k] = sum c^2 (f32 exact).
__global__ __launch_bounds__(256) void vq_prep_cb(const float* __restrict__ cb,
                                                  _Float16* __restrict__ cb16,
                                                  float* __restrict__ cnorm) {
    __shared__ float red[256];
    int k = blockIdx.x, t = threadIdx.x;
    float v = cb[(size_t)k * DIMS + t];
    cb16[(size_t)k * DIMS + t] = (_Float16)v;
    red[t] = v * v;
    __syncthreads();
    for (int s = 128; s > 0; s >>= 1) {
        if (t < s) red[t] += red[t + s];
        __syncthreads();
    }
    if (t == 0) cnorm[k] = red[0];
}

// ---------------------------------------------------------------------------
// z: (b, d, hw) f32 -> z16: (n=b*1024+hw, d) f16 via LDS transpose tile
// (64 hw x 256 d per block), plus znorm[n] = sum_d z^2 in exact f32.
__global__ __launch_bounds__(256) void vq_prep_z(const float* __restrict__ z,
                                                 _Float16* __restrict__ z16,
                                                 float* __restrict__ znorm) {
    __shared__ float tile[256][65];   // [dd][hw], pad -> conflict-free transpose
    __shared__ float nred[4][64];
    int b      = blockIdx.x >> 4;
    int hwbase = (blockIdx.x & 15) << 6;
    int t = threadIdx.x;
    int hw_l = t & 63;
    int dd4  = t >> 6;                // 0..3
    const float* zb = z + (size_t)b * (DIMS * HW);
    float nacc = 0.0f;
    #pragma unroll 4
    for (int i = 0; i < 64; ++i) {
        int dd = i * 4 + dd4;
        float v = zb[(size_t)dd * HW + hwbase + hw_l];  // coalesced over hw
        tile[dd][hw_l] = v;
        nacc += v * v;
    }
    nred[dd4][hw_l] = nacc;
    __syncthreads();
    if (t < 64) {
        znorm[b * HW + hwbase + t] =
            nred[0][t] + nred[1][t] + nred[2][t] + nred[3][t];
    }
    // write z16[n][dd]: consecutive t -> consecutive dd (coalesced 512B rows)
    for (int j = 0; j < 64; ++j) {
        z16[((size_t)(b * HW + hwbase + j)) * DIMS + t] = (_Float16)tile[t][j];
    }
}

// ---------------------------------------------------------------------------
// Main fused kernel: one block per 16-row strip of z_flat. 8 waves; wave w
// owns k-tiles {w*16 + 128*i}. Per tile: 8 chained WMMA f32_16x16x32_f16 over
// d=256, emit distances, track running argmin; shfl+LDS reduce to indices.
__global__ __launch_bounds__(256) void vq_gemm_argmin(
    const _Float16* __restrict__ z16, const _Float16* __restrict__ cb16,
    const float* __restrict__ znorm, const float* __restrict__ cnorm,
    float* __restrict__ dist, float* __restrict__ idxf, int* __restrict__ idxi) {
    __shared__ float red_min[8][16];
    __shared__ int   red_idx[8][16];

    const int nbase  = blockIdx.x << 4;
    const int tid    = threadIdx.x;
    const int wave   = tid >> 5;
    const int lane   = tid & 31;
    const int lane16 = lane & 15;
    const int hi     = (lane >= 16) ? 1 : 0;

    // --- A fragments (16x32 f16): lane m = lane16,
    //     halves 0..15 = K{base+0..15 | +16..31} (contiguous, matches B)
    const _Float16* zrow = z16 + (size_t)(nbase + lane16) * DIMS;
    half16 afrag[8];
    #pragma unroll
    for (int dt = 0; dt < 8; ++dt) {
        int dlo = dt * 32 + hi * 16;
        half8 alo = *(const half8*)(zrow + dlo);
        half8 ahi = *(const half8*)(zrow + dlo + 8);
        afrag[dt] = __builtin_shufflevector(alo, ahi,
                        0,1,2,3,4,5,6,7,8,9,10,11,12,13,14,15);
    }
    float zn[8];
    #pragma unroll
    for (int r = 0; r < 8; ++r) zn[r] = znorm[nbase + r + hi * 8];

    float minv[8]; int mink[8];
    #pragma unroll
    for (int r = 0; r < 8; ++r) { minv[r] = 3.4e38f; mink[r] = 0; }

    for (int kbase = wave * 16; kbase < KCB; kbase += 128) {
        // B fragment (32x16 f16): lane n = lane16, 16 contiguous K halves
        const _Float16* crow = cb16 + (size_t)(kbase + lane16) * DIMS;
        float cn = cnorm[kbase + lane16];
        float8 acc = {0.f, 0.f, 0.f, 0.f, 0.f, 0.f, 0.f, 0.f};
        #pragma unroll
        for (int dt = 0; dt < 8; ++dt) {
            half16 bfrag = *(const half16*)(crow + dt * 32 + hi * 16);
            acc = __builtin_amdgcn_wmma_f32_16x16x32_f16(
                      false, afrag[dt], false, bfrag,
                      (short)0, acc, false, false);
        }
        const int kc = kbase + lane16;
        #pragma unroll
        for (int r = 0; r < 8; ++r) {
            // D layout: vgpr r -> row M = r (+8 for upper half-wave), col = lane16
            float dv = zn[r] + cn - 2.0f * acc[r];
            dist[(size_t)(nbase + r + hi * 8) * KCB + kc] = dv;
            if (dv < minv[r] || (dv == minv[r] && kc < mink[r])) {
                minv[r] = dv; mink[r] = kc;
            }
        }
    }
    // reduce argmin across the 16 lanes of each half-wave (rows differ by half)
    #pragma unroll
    for (int m = 1; m < 16; m <<= 1) {
        #pragma unroll
        for (int r = 0; r < 8; ++r) {
            float ov = __shfl_xor(minv[r], m, 32);
            int   ok = __shfl_xor(mink[r], m, 32);
            if (ov < minv[r] || (ov == minv[r] && ok < mink[r])) {
                minv[r] = ov; mink[r] = ok;
            }
        }
    }
    if (lane16 == 0) {
        #pragma unroll
        for (int r = 0; r < 8; ++r) {
            red_min[wave][r + hi * 8] = minv[r];
            red_idx[wave][r + hi * 8] = mink[r];
        }
    }
    __syncthreads();
    if (tid < 16) {   // final reduce across the 8 waves' k-subsets
        float bv = red_min[0][tid]; int bk = red_idx[0][tid];
        #pragma unroll
        for (int w = 1; w < 8; ++w) {
            float v = red_min[w][tid]; int k = red_idx[w][tid];
            if (v < bv || (v == bv && k < bk)) { bv = v; bk = k; }
        }
        idxi[nbase + tid] = bk;
        idxf[nbase + tid] = (float)bk;
    }
}

// ---------------------------------------------------------------------------
// quantized_st (== zq in forward) in (b,d,h,w) layout + SSE accumulation.
__global__ __launch_bounds__(256) void vq_gather_mse(
    const float* __restrict__ z, const float* __restrict__ cb,
    const int* __restrict__ idx, float* __restrict__ outq,
    float* __restrict__ sse) {
    __shared__ float red[256];
    const size_t total  = (size_t)BATCH * DIMS * HW;   // 8,388,608
    const size_t stride = (size_t)gridDim.x * blockDim.x;
    float acc = 0.0f;
    for (size_t i = (size_t)blockIdx.x * blockDim.x + threadIdx.x;
         i < total; i += stride) {
        int b   = (int)(i >> 18);
        int rem = (int)(i & 262143);
        int dd  = rem >> 10;
        int hw  = rem & 1023;
        int n   = (b << 10) | hw;
        float q = cb[(size_t)idx[n] * DIMS + dd];  // codebook L2-resident (1 MB)
        outq[i] = q;
        float d = q - z[i];
        acc += d * d;
    }
    red[threadIdx.x] = acc;
    __syncthreads();
    for (int s = 128; s > 0; s >>= 1) {
        if (threadIdx.x < s) red[threadIdx.x] += red[threadIdx.x + s];
        __syncthreads();
    }
    if (threadIdx.x == 0) atomicAdd(sse, red[0]);   // global_atomic_add_f32
}

// ---------------------------------------------------------------------------
__global__ void vq_hist(const int* __restrict__ idx, float* __restrict__ hist) {
    int i = blockIdx.x * blockDim.x + threadIdx.x;
    if (i < N_ROWS) atomicAdd(&hist[idx[i]], 1.0f);
}

// ---------------------------------------------------------------------------
// loss = 1.25 * MSE ; perplexity = exp(-sum p*log(p+eps))
__global__ __launch_bounds__(1024) void vq_final(const float* __restrict__ hist,
                                                 const float* __restrict__ sse,
                                                 float* __restrict__ out_scalars) {
    __shared__ float red[1024];
    int t = threadIdx.x;
    float p = hist[t] * (1.0f / (float)N_ROWS);
    red[t] = p * logf(p + 1e-10f);
    __syncthreads();
    for (int s = 512; s > 0; s >>= 1) {
        if (t < s) red[t] += red[t + s];
        __syncthreads();
    }
    if (t == 0) {
        out_scalars[0] = (*sse) * (1.25f / (float)((size_t)BATCH * DIMS * HW));
        out_scalars[1] = expf(-red[0]);
    }
}

// ---------------------------------------------------------------------------
extern "C" void kernel_launch(void* const* d_in, const int* in_sizes, int n_in,
                              void* d_out, int out_size, void* d_ws, size_t ws_size,
                              hipStream_t stream) {
    const float* z  = (const float*)d_in[0];   // (32,256,32,32)
    const float* cb = (const float*)d_in[1];   // (1024,256)

    float* out         = (float*)d_out;
    float* outq        = out;                  // 8,388,608 quantized_st
    float* out_scalars = out + 8388608;        // [loss, perplexity]
    float* out_idxf    = out + 8388610;        // 32,768 indices (as float)
    float* out_dist    = out + 8421378;        // 33,554,432 distances

    char* ws = (char*)d_ws;
    _Float16* z16  = (_Float16*)ws;  ws += (size_t)N_ROWS * DIMS * 2;  // 16 MB
    _Float16* cb16 = (_Float16*)ws;  ws += (size_t)KCB    * DIMS * 2;  // 512 KB
    float* znorm = (float*)ws;       ws += (size_t)N_ROWS * 4;
    float* cnorm = (float*)ws;       ws += (size_t)KCB * 4;
    int*   idxi  = (int*)ws;         ws += (size_t)N_ROWS * 4;
    float* hist  = (float*)ws;       ws += (size_t)KCB * 4;
    float* sse   = (float*)ws;

    vq_init<<<1, 1024, 0, stream>>>(hist, sse);
    vq_prep_cb<<<KCB, 256, 0, stream>>>(cb, cb16, cnorm);
    vq_prep_z<<<512, 256, 0, stream>>>(z, z16, znorm);
    vq_gemm_argmin<<<N_ROWS / 16, 256, 0, stream>>>(z16, cb16, znorm, cnorm,
                                                    out_dist, out_idxf, idxi);
    vq_gather_mse<<<4096, 256, 0, stream>>>(z, cb, idxi, outq, sse);
    vq_hist<<<N_ROWS / 256, 256, 0, stream>>>(idxi, hist);
    vq_final<<<1, 1024, 0, stream>>>(hist, sse, out_scalars);
}
